// MMD_loss_31396210933816
// MI455X (gfx1250) — matrix operand order repaired and checked by reading, not compiled, not run on hardware
//
#include <hip/hip_runtime.h>
#include <hip/hip_bf16.h>

typedef __bf16 bf16_t;
typedef __attribute__((ext_vector_type(16))) __bf16 v16bf;
typedef __attribute__((ext_vector_type(8)))  __bf16 v8bf;
typedef __attribute__((ext_vector_type(4)))  __bf16 v4bf;
typedef __attribute__((ext_vector_type(8)))  float  v8f;

#define N_TOT   8192
#define HALF_N  4096
#define DIMK    1024
#define ROW_BYTES 2048u                       // 1024 bf16 * 2B
// ws layout: [hi bf16 8M][lo bf16 8M][float region]
#define WS_F32_BYTE_OFF (2u * 8388608u * 2u)  // 33,554,432 bytes
// float region indices: sq[0..8191], colpart[8192..16383], band0[16384], acc[16385]

union ABFrag { v16bf v; v8bf h[2]; };

__device__ __forceinline__ v8bf ld8bf(const bf16_t* base, uint32_t byteoff) {
    return *reinterpret_cast<const v8bf*>(
        reinterpret_cast<const char*>(base) + byteoff);
}

// ---------------- K0: split fp32 -> bf16 hi + bf16 lo --------------------
__global__ __launch_bounds__(256) void k_split(const float* __restrict__ src,
                                               const float* __restrict__ tgt,
                                               bf16_t* __restrict__ hi,
                                               bf16_t* __restrict__ lo) {
    size_t g4  = (size_t)blockIdx.x * blockDim.x + threadIdx.x;  // float4 index
    size_t idx = g4 * 4;                                          // element index
    const float* p = (idx < (size_t)HALF_N * DIMK)
                       ? (src + idx)
                       : (tgt + (idx - (size_t)HALF_N * DIMK));
    float4 v = *reinterpret_cast<const float4*>(p);
    float xs[4] = {v.x, v.y, v.z, v.w};
    v4bf h4, l4;
#pragma unroll
    for (int e = 0; e < 4; ++e) {
        bf16_t hb = (bf16_t)xs[e];
        float  r  = xs[e] - (float)hb;
        h4[e] = hb;
        l4[e] = (bf16_t)r;
    }
    *reinterpret_cast<v4bf*>(hi + idx) = h4;
    *reinterpret_cast<v4bf*>(lo + idx) = l4;
}

// ---------------- K1: per-row squared L2 norms ---------------------------
__global__ __launch_bounds__(256) void k_sq(const float* __restrict__ src,
                                            const float* __restrict__ tgt,
                                            float* __restrict__ sq) {
    int row = blockIdx.x;
    const float* p = (row < HALF_N) ? (src + (size_t)row * DIMK)
                                    : (tgt + (size_t)(row - HALF_N) * DIMK);
    float4 v = *reinterpret_cast<const float4*>(p + threadIdx.x * 4);
    float s = v.x * v.x + v.y * v.y + v.z * v.z + v.w * v.w;
    __shared__ float red[256];
    red[threadIdx.x] = s;
    __syncthreads();
    for (int off = 128; off > 0; off >>= 1) {
        if (threadIdx.x < off) red[threadIdx.x] += red[threadIdx.x + off];
        __syncthreads();
    }
    if (threadIdx.x == 0) sq[row] = red[0];
}

// ---------------- K2: partial column sums (8 row-chunks x 1024 cols) -----
__global__ __launch_bounds__(256) void k_colsum(const float* __restrict__ src,
                                                const float* __restrict__ tgt,
                                                float* __restrict__ colpart) {
    int d     = blockIdx.x * 256 + threadIdx.x;   // column 0..1023
    int chunk = blockIdx.y;                        // 0..7 -> rows chunk*1024..
    const float* base = (chunk < 4)
        ? (src + (size_t)chunk * 1024 * DIMK)
        : (tgt + (size_t)(chunk - 4) * 1024 * DIMK);
    float s = 0.0f;
    for (int r = 0; r < 1024; ++r)
        s += base[(size_t)r * DIMK + d];
    colpart[chunk * DIMK + d] = s;
}

// ---------------- K3: bandwidth via closed form; zero accumulator --------
__global__ __launch_bounds__(256) void k_band(const float* __restrict__ sq,
                                              const float* __restrict__ colpart,
                                              float* __restrict__ band0,
                                              float* __restrict__ acc) {
    int tid = threadIdx.x;
    float s1 = 0.0f;
    for (int i = tid; i < N_TOT; i += 256) s1 += sq[i];
    float s2 = 0.0f;
    for (int d = tid; d < DIMK; d += 256) {
        float cs = 0.0f;
#pragma unroll
        for (int c = 0; c < 8; ++c) cs += colpart[c * DIMK + d];
        s2 += cs * cs;
    }
    __shared__ float r1[256], r2[256];
    r1[tid] = s1; r2[tid] = s2;
    __syncthreads();
    for (int off = 128; off > 0; off >>= 1) {
        if (tid < off) { r1[tid] += r1[tid + off]; r2[tid] += r2[tid + off]; }
        __syncthreads();
    }
    if (tid == 0) {
        float sumL2 = 2.0f * (float)N_TOT * r1[0] - 2.0f * r2[0];
        float band  = sumL2 / ((float)N_TOT * (float)N_TOT - (float)N_TOT);
        band0[0] = band * 0.25f;   // / KERNEL_MUL^(KERNEL_NUM//2) = /4
        acc[0]   = 0.0f;           // zero the signed quadrant accumulator
    }
}

// ---------------- K4: fused WMMA Gram -> L2 -> 5-exp -> signed sum -------
__global__ __launch_bounds__(256) void k_mmd(const bf16_t* __restrict__ hi,
                                             const bf16_t* __restrict__ lo,
                                             const float* __restrict__ sq,
                                             const float* __restrict__ band0p,
                                             float* __restrict__ acc) {
    const int tile_i = blockIdx.y;                 // 0..63 (rows, 128 each)
    const int tile_j = blockIdx.x;                 // 0..63 (cols, 128 each)
    const int wave   = threadIdx.x >> 5;           // 0..7
    const int lane   = threadIdx.x & 31;
    const int l      = lane & 15;
    const int hh     = lane >> 4;                  // lane half
    const int row_base = tile_i * 128 + wave * 16;
    const int col_base = tile_j * 128;

    // 32-bit byte offsets (arrays are 16.7 MB -> fits u32; enables saddr form)
    const uint32_t arow = (uint32_t)(row_base + l) * ROW_BYTES + (uint32_t)(16 * hh);
    uint32_t brow[8];
#pragma unroll
    for (int j = 0; j < 8; ++j)
        brow[j] = (uint32_t)(col_base + j * 16 + l) * ROW_BYTES + (uint32_t)(16 * hh);

    v8f acc8[8] = {};

    for (int kc = 0; kc < DIMK; kc += 32) {
        const uint32_t kb = (uint32_t)kc * 2u;     // byte offset of K chunk
        // A fragments (16x32 bf16): lane l = row M=l; half selects K grouping
        ABFrag ahi, alo;
        ahi.h[0] = ld8bf(hi, arow + kb);
        ahi.h[1] = ld8bf(hi, arow + kb + 32u);     // K + 16 elements
        alo.h[0] = ld8bf(lo, arow + kb);
        alo.h[1] = ld8bf(lo, arow + kb + 32u);

#pragma unroll
        for (int jg = 0; jg < 8; jg += 4) {
            // B fragments (32x16 bf16): lane l = col N=l; half selects K grouping
            ABFrag bhi[4], blo[4];
#pragma unroll
            for (int u = 0; u < 4; ++u) {
                const uint32_t bo = brow[jg + u] + kb;
                bhi[u].h[0] = ld8bf(hi, bo);
                bhi[u].h[1] = ld8bf(hi, bo + 32u);  // K + 16 elements
                blo[u].h[0] = ld8bf(lo, bo);
                blo[u].h[1] = ld8bf(lo, bo + 32u);
            }
            // bf16x3, term-major: consecutive WMMAs hit different accumulators,
            // breaking the D->C chain (bf16 WMMA->WMMA hazard = 5 slots)
#pragma unroll
            for (int u = 0; u < 4; ++u)
                acc8[jg + u] = __builtin_amdgcn_wmma_f32_16x16x32_bf16(
                    false, ahi.v, false, bhi[u].v, (short)0, acc8[jg + u], false, false);
#pragma unroll
            for (int u = 0; u < 4; ++u)
                acc8[jg + u] = __builtin_amdgcn_wmma_f32_16x16x32_bf16(
                    false, ahi.v, false, blo[u].v, (short)0, acc8[jg + u], false, false);
#pragma unroll
            for (int u = 0; u < 4; ++u)
                acc8[jg + u] = __builtin_amdgcn_wmma_f32_16x16x32_bf16(
                    false, alo.v, false, bhi[u].v, (short)0, acc8[jg + u], false, false);
        }
    }

    // Epilogue: L2 = sq_i + sq_j - 2G ; sum of 5 exp kernels
    const float band0 = band0p[0];
    const float inv0  = 1.0f / band0;
    float rsq[8];
#pragma unroll
    for (int r = 0; r < 8; ++r) rsq[r] = sq[row_base + 8 * hh + r];

    float partial = 0.0f;
#pragma unroll
    for (int j = 0; j < 8; ++j) {
        float csq = sq[col_base + j * 16 + l];
#pragma unroll
        for (int r = 0; r < 8; ++r) {
            float L2v = rsq[r] + csq - 2.0f * acc8[j][r];
            float f = inv0, s = 0.0f;
#pragma unroll
            for (int t = 0; t < 5; ++t) { s += __expf(-L2v * f); f *= 0.5f; }
            partial += s;
        }
    }

    __shared__ float red[256];
    red[threadIdx.x] = partial;
    __syncthreads();
    for (int off = 128; off > 0; off >>= 1) {
        if (threadIdx.x < off) red[threadIdx.x] += red[threadIdx.x + off];
        __syncthreads();
    }
    if (threadIdx.x == 0) {
        float sign = ((tile_i < 32) == (tile_j < 32)) ? 1.0f : -1.0f;
        atomicAdd(acc, sign * red[0]);
    }
}

// ---------------- K5: scale to means and write scalar --------------------
__global__ void k_final(const float* __restrict__ acc, float* __restrict__ out) {
    out[0] = acc[0] * (1.0f / ((float)HALF_N * (float)HALF_N));
}

extern "C" void kernel_launch(void* const* d_in, const int* in_sizes, int n_in,
                              void* d_out, int out_size, void* d_ws, size_t ws_size,
                              hipStream_t stream) {
    const float* src = (const float*)d_in[0];
    const float* tgt = (const float*)d_in[1];
    bf16_t* hi  = (bf16_t*)d_ws;
    bf16_t* lo  = hi + (size_t)N_TOT * DIMK;
    float*  wsf = (float*)((char*)d_ws + WS_F32_BYTE_OFF);
    float*  sq      = wsf;
    float*  colpart = wsf + 8192;
    float*  band0   = wsf + 16384;
    float*  acc     = wsf + 16385;
    float*  out     = (float*)d_out;

    k_split <<<8192, 256, 0, stream>>>(src, tgt, hi, lo);
    k_sq    <<<8192, 256, 0, stream>>>(src, tgt, sq);
    k_colsum<<<dim3(4, 8), 256, 0, stream>>>(src, tgt, colpart);
    k_band  <<<1, 256, 0, stream>>>(sq, colpart, band0, acc);
    k_mmd   <<<dim3(64, 64), 256, 0, stream>>>(hi, lo, sq, band0, acc);
    k_final <<<1, 1, 0, stream>>>(acc, out);
}
